// FullyHyperbolicCNN_Lorentz_60619168415850
// MI455X (gfx1250) — matrix operand order repaired and checked by reading, not compiled, not run on hardware
//
#include <hip/hip_runtime.h>

typedef __attribute__((ext_vector_type(16))) _Float16 v16h;
typedef __attribute__((ext_vector_type(8)))  float    v8f;

#define EPS   1e-6f
#define BNEPS 1e-5f

// ---------------- small utility kernels ----------------

__global__ void k_zero(float* p, int n) {
    int i = blockIdx.x * blockDim.x + threadIdx.x;
    if (i < n) p[i] = 0.f;
}

__global__ void k_cvt_f16(const float* __restrict__ src, _Float16* __restrict__ dst, int n) {
    int i = blockIdx.x * blockDim.x + threadIdx.x;
    if (i < n) dst[i] = (_Float16)src[i];
}

// finalize BN: st = [sum(nc)][sumsq(nc)][scale(nc)][shift(nc)]
__global__ void k_bn_fin(float* st, const float* __restrict__ g, const float* __restrict__ b,
                         float n, int nc) {
    int c = threadIdx.x;
    if (c >= nc) return;
    float mean = st[c] / n;
    float var  = st[nc + c] / n - mean * mean;
    float sc   = g[c] * rsqrtf(var + BNEPS);
    st[2 * nc + c] = sc;
    st[3 * nc + c] = b[c] - mean * sc;
}

// ---------------- stage 1: conv1 + Lorentz epilogue + relu ----------------
// x:(B,28,28,1)  W1:(3,3,1,32)  v1:(32)  out s1:(B,14,14,32) f16 (pre-BN, post-relu)
__global__ void __launch_bounds__(256) k_conv1(const float* __restrict__ x,
                                               const float* __restrict__ w1,
                                               const float* __restrict__ v1,
                                               _Float16* __restrict__ s1, int total) {
    int m = blockIdx.x * blockDim.x + threadIdx.x;
    if (m >= total) return;
    int b = m / 196, pix = m % 196, oh = pix / 14, ow = pix % 14;

    float p[9];
    float patch2 = 0.f;
#pragma unroll
    for (int kh = 0; kh < 3; ++kh)
#pragma unroll
        for (int kw = 0; kw < 3; ++kw) {
            int ih = 2 * oh + kh, iw = 2 * ow + kw;       // SAME: pad_low=0
            float v = (ih < 28 && iw < 28) ? x[b * 784 + ih * 28 + iw] : 0.f;
            p[kh * 3 + kw] = v;
            patch2 += v * v;
        }

    float v2 = 0.f;
#pragma unroll 8
    for (int c = 0; c < 32; ++c) { float vv = v1[c]; v2 += vv * vv; }
    float gmm = rsqrtf(1.f - v2);

    float y[32];
    float y2 = 0.f;
#pragma unroll
    for (int c = 0; c < 32; ++c) {
        float acc = 0.f;
#pragma unroll
        for (int i = 0; i < 9; ++i) acc = fmaf(p[i], w1[i * 32 + c], acc);
        y[c] = acc;
        y2 += acc * acc;
    }
    float f = sqrtf((patch2 + EPS) / (y2 + EPS));
    float t = sqrtf(patch2 + 1.f);
    float vs = 0.f;
#pragma unroll
    for (int c = 0; c < 32; ++c) { y[c] *= f; vs = fmaf(y[c], v1[c], vs); }
    float coef = (gmm - 1.f) / (v2 + EPS) * vs + gmm * t;
#pragma unroll
    for (int c = 0; c < 32; ++c) {
        float s = fmaxf(fmaf(coef, v1[c], y[c]), 0.f);    // boost + relu
        s1[(size_t)m * 32 + c] = (_Float16)s;
    }
}

// ---------------- BN stats (f16 src, 32 ch / f32 src, 64 ch) ----------------
__global__ void k_stats1(const _Float16* __restrict__ s1, float* st, int pixCount) {
    int gid = blockIdx.x * blockDim.x + threadIdx.x;
    int c = gid & 31, p0 = gid >> 5;
    int stride = (gridDim.x * blockDim.x) >> 5;
    float s = 0.f, sq = 0.f;
    for (int p = p0; p < pixCount; p += stride) {
        float v = (float)s1[(size_t)p * 32 + c];
        s += v; sq += v * v;
    }
    atomicAdd(&st[c], s);
    atomicAdd(&st[32 + c], sq);
}

__global__ void k_stats2(const float* __restrict__ s2, float* st, int pixCount) {
    int gid = blockIdx.x * blockDim.x + threadIdx.x;
    int c = gid & 63, p0 = gid >> 6;
    int stride = (gridDim.x * blockDim.x) >> 6;
    float s = 0.f, sq = 0.f;
    for (int p = p0; p < pixCount; p += stride) {
        float v = s2[(size_t)p * 64 + c];
        s += v; sq += v * v;
    }
    atomicAdd(&st[c], s);
    atomicAdd(&st[64 + c], sq);
}

// ---------------- stage 2: conv2 as implicit-GEMM WMMA + Lorentz epilogue ----------------
// Per wave: 16(M)x64(N) tile, K-loop = 9 taps x 32 ch. BN1 affine fused into A-fragment.
// stats layout: scale1 at st[64..95], shift1 at st[96..127].
__global__ void __launch_bounds__(128) k_conv2_wmma(const _Float16* __restrict__ s1,
                                                    const _Float16* __restrict__ w2h,
                                                    const float* __restrict__ st,
                                                    const float* __restrict__ v64,
                                                    float* __restrict__ s2) {
    const int lane  = threadIdx.x & 31;
    const int wv    = threadIdx.x >> 5;
    const int mBase = (blockIdx.x * 4 + wv) * 16;
    const int mA    = mBase + (lane & 15);                // A-matrix row for this lane
    const int b  = mA / 49, pix = mA % 49;
    const int oh = pix / 7, ow = pix % 7;
    const int kofl = (lane & 16) ? 8 : 0;                 // A K-offset per ISA layout
    const int kbB  = (lane & 16) ? 8 : 0;                 // B K-offset per ISA layout

    // fused BN1 affine coefficients for this lane's 16 A K-slots (channels)
    float scA[16], shA[16];
#pragma unroll
    for (int e = 0; e < 16; ++e) {
        int g = e >> 1;
        int ch = kofl + ((g < 4) ? 2 * g : 16 + 2 * (g - 4)) + (e & 1);
        scA[e] = st[64 + ch];
        shA[e] = st[96 + ch];
    }
    float vN[4];
#pragma unroll
    for (int nt = 0; nt < 4; ++nt) vN[nt] = v64[nt * 16 + (lane & 15)];
    float v2 = 0.f;
#pragma unroll 8
    for (int c = 0; c < 64; ++c) { float vv = v64[c]; v2 += vv * vv; }
    const float gmm = rsqrtf(1.f - v2);

    __builtin_prefetch(s1 + ((size_t)(b * 14 + 2 * oh) * 14 + 2 * ow) * 32, 0, 3);

    v8f acc[4] = {};
    float patch2 = 0.f;

#pragma unroll
    for (int tap = 0; tap < 9; ++tap) {
        const int kh = tap / 3, kw = tap % 3;
        const int ih = 2 * oh + kh, iw = 2 * ow + kw;     // SAME: pad_low=0
        const bool inb = (ih < 14) && (iw < 14);
        const _Float16* ap = s1 + ((size_t)(b * 14 + ih) * 14 + iw) * 32;

        v16h afr;
#pragma unroll
        for (int g = 0; g < 8; ++g) {
            const int kk = kofl + ((g < 4) ? 2 * g : 16 + 2 * (g - 4));
            float a0 = 0.f, a1 = 0.f;
            if (inb) {
                a0 = fmaf((float)ap[kk],     scA[2 * g],     shA[2 * g]);
                a1 = fmaf((float)ap[kk + 1], scA[2 * g + 1], shA[2 * g + 1]);
            }
            patch2 += a0 * a0 + a1 * a1;                  // sum over all K for this row
            afr[2 * g]     = (_Float16)a0;
            afr[2 * g + 1] = (_Float16)a1;
        }

        const _Float16* wp = w2h + (size_t)tap * 32 * 64;
#pragma unroll
        for (int nt = 0; nt < 4; ++nt) {
            const int co = nt * 16 + (lane & 15);
            v16h bfr;
#pragma unroll
            for (int j = 0; j < 8; ++j) {
                const int ci = kbB + ((j < 4) ? 2 * j : 16 + 2 * (j - 4));
                bfr[2 * j]     = wp[ci * 64 + co];
                bfr[2 * j + 1] = wp[(ci + 1) * 64 + co];
            }
            acc[nt] = __builtin_amdgcn_wmma_f32_16x16x32_f16(
                false, afr, false, bfr, (short)0, acc[nt], false, false);
        }
    }

    // row M=lane&15 patch2 lives split across lane pair (l, l^16): combine
    patch2 += __shfl_xor(patch2, 16, 32);

    // C/D layout: lane = N (mod 16), VGPR r = row r (lanes<16) / 8+r (lanes>=16)
    float y2r[8], p2r[8], fr[8];
#pragma unroll
    for (int r = 0; r < 8; ++r) {
        float s = acc[0][r] * acc[0][r] + acc[1][r] * acc[1][r] +
                  acc[2][r] * acc[2][r] + acc[3][r] * acc[3][r];
        s += __shfl_xor(s, 1, 32); s += __shfl_xor(s, 2, 32);
        s += __shfl_xor(s, 4, 32); s += __shfl_xor(s, 8, 32);
        y2r[r] = s;
        p2r[r] = __shfl(patch2, ((lane & 16) ? 8 : 0) + r, 32);
        fr[r]  = sqrtf((p2r[r] + EPS) / (y2r[r] + EPS));
    }
    float vsr[8];
#pragma unroll
    for (int r = 0; r < 8; ++r) {
        float vs = 0.f;
#pragma unroll
        for (int nt = 0; nt < 4; ++nt) {
            acc[nt][r] *= fr[r];                          // rescale
            vs = fmaf(acc[nt][r], vN[nt], vs);
        }
        vs += __shfl_xor(vs, 1, 32); vs += __shfl_xor(vs, 2, 32);
        vs += __shfl_xor(vs, 4, 32); vs += __shfl_xor(vs, 8, 32);
        vsr[r] = vs;
    }
    const float inv = (gmm - 1.f) / (v2 + EPS);
#pragma unroll
    for (int r = 0; r < 8; ++r) {
        const float coef = fmaf(inv, vsr[r], gmm * sqrtf(p2r[r] + 1.f));
        const int mg = mBase + ((lane & 16) ? 8 : 0) + r;
#pragma unroll
        for (int nt = 0; nt < 4; ++nt) {
            float sv = fmaxf(fmaf(coef, vN[nt], acc[nt][r]), 0.f);   // boost + relu
            s2[(size_t)mg * 64 + nt * 16 + (lane & 15)] = sv;        // coalesced across lanes
        }
    }
}

// ---------------- head: pool + BN2 affine + project + lin + relu + MLR ----------------
// stats: scale2 at st[256..319], shift2 at st[320..383]
__global__ void __launch_bounds__(256) k_head(const float* __restrict__ s2,
                                              const float* __restrict__ st,
                                              const float* __restrict__ linw,
                                              const float* __restrict__ mlrw,
                                              float* __restrict__ out) {
    __shared__ float buf[8 * 64];
    const int lane = threadIdx.x & 31, wv = threadIdx.x >> 5;
    const int b = blockIdx.x * 8 + wv;
    const float* base = s2 + (size_t)b * 49 * 64;

    float m0 = 0.f, m1 = 0.f;
    for (int p = 0; p < 49; ++p) {
        m0 += base[p * 64 + lane];
        m1 += base[p * 64 + 32 + lane];
    }
    m0 *= (1.f / 49.f); m1 *= (1.f / 49.f);
    m0 = fmaf(m0, st[256 + lane],      st[320 + lane]);          // BN2 (affine commutes w/ mean)
    m1 = fmaf(m1, st[256 + 32 + lane], st[320 + 32 + lane]);
    buf[wv * 64 + lane] = m0;
    buf[wv * 64 + 32 + lane] = m1;
    __syncthreads();

    float sn = m0 * m0 + m1 * m1;
    sn += __shfl_xor(sn, 1, 32); sn += __shfl_xor(sn, 2, 32);
    sn += __shfl_xor(sn, 4, 32); sn += __shfl_xor(sn, 8, 32); sn += __shfl_xor(sn, 16, 32);
    const float t0 = sqrtf(sn + 1.f);                            // project time channel

    float q0 = linw[lane * 65] * t0;
    float q1 = linw[(lane + 32) * 65] * t0;
    for (int k = 0; k < 64; ++k) {
        const float mk = buf[wv * 64 + k];
        q0 = fmaf(linw[lane * 65 + 1 + k], mk, q0);
        q1 = fmaf(linw[(lane + 32) * 65 + 1 + k], mk, q1);
    }
    const float r0 = fmaxf(q0, 0.f), r1 = fmaxf(q1, 0.f);
    float rn = r0 * r0 + r1 * r1;
    rn += __shfl_xor(rn, 1, 32); rn += __shfl_xor(rn, 2, 32);
    rn += __shfl_xor(rn, 4, 32); rn += __shfl_xor(rn, 8, 32); rn += __shfl_xor(rn, 16, 32);
    const float t2 = sqrtf(rn + 1.f);

    __syncthreads();
    buf[wv * 64 + lane] = r0;
    buf[wv * 64 + 32 + lane] = r1;
    __syncthreads();

    if (lane < 10) {
        float o = -t2 * mlrw[lane * 65];                         // signs flip time channel
        for (int j = 0; j < 64; ++j) o = fmaf(buf[wv * 64 + j], mlrw[lane * 65 + 1 + j], o);
        out[b * 10 + lane] = o;
    }
}

// ---------------- launch ----------------

extern "C" void kernel_launch(void* const* d_in, const int* in_sizes, int n_in,
                              void* d_out, int out_size, void* d_ws, size_t ws_size,
                              hipStream_t stream) {
    const float* x   = (const float*)d_in[0];
    const float* w1  = (const float*)d_in[1];
    const float* v1  = (const float*)d_in[2];
    const float* w2  = (const float*)d_in[3];
    const float* v2c = (const float*)d_in[4];
    const float* g1  = (const float*)d_in[5];
    const float* b1  = (const float*)d_in[6];
    const float* g2  = (const float*)d_in[7];
    const float* b2  = (const float*)d_in[8];
    const float* lw  = (const float*)d_in[9];
    const float* mw  = (const float*)d_in[10];
    float* out = (float*)d_out;

    char* ws = (char*)d_ws;
    float*    stats = (float*)ws;                              // 384 floats (sums/scale/shift x2)
    _Float16* w2h   = (_Float16*)(ws + 2048);                  // 18432 f16
    _Float16* s1    = (_Float16*)(ws + 40960);                 // 4096*196*32 f16
    float*    s2    = (float*)(ws + 40960 + 51380224ull);      // 4096*49*64 f32

    const int P1 = 4096 * 196;   // 802816 stage-1 pixels
    const int P2 = 4096 * 49;    // 200704 stage-2 pixels

    k_zero    <<<2,    256, 0, stream>>>(stats, 384);
    k_cvt_f16 <<<72,   256, 0, stream>>>(w2, w2h, 9 * 32 * 64);
    k_conv1   <<<3136, 256, 0, stream>>>(x, w1, v1, s1, P1);
    k_stats1  <<<256,  256, 0, stream>>>(s1, stats, P1);
    k_bn_fin  <<<1,    64,  0, stream>>>(stats, g1, b1, (float)P1, 32);
    k_conv2_wmma<<<3136, 128, 0, stream>>>(s1, w2h, stats, v2c, s2);
    k_stats2  <<<256,  256, 0, stream>>>(s2, stats + 128, P2);
    k_bn_fin  <<<1,    64,  0, stream>>>(stats + 128, g2, b2, (float)P2, 64);
    k_head    <<<512,  256, 0, stream>>>(s2, stats, lw, mw, out);
}
